// AttentionFreeTransformer_17952963297387
// MI455X (gfx1250) — matrix operand order repaired and checked by reading, not compiled, not run on hardware
//
#include <hip/hip_runtime.h>

#define BB 32
#define NN 1024
#define DD 512
#define LP 40   // LDS row pitch in bf16 elements (80B: 16B-aligned rows, conflict-free frag loads)

typedef __bf16 bf16;
typedef bf16  v16bf __attribute__((ext_vector_type(16)));
typedef bf16  v8bf  __attribute__((ext_vector_type(8)));
typedef bf16  v4bf  __attribute__((ext_vector_type(4)));
typedef float v8f   __attribute__((ext_vector_type(8)));
typedef float v4f   __attribute__((ext_vector_type(4)));

__device__ __forceinline__ v16bf frag_cat(v8bf lo, v8bf hi) {
  v16bf r;
#pragma unroll
  for (int i = 0; i < 8; ++i) { r[i] = lo[i]; r[8 + i] = hi[i]; }
  return r;
}

// A fragment (16x32 bf16, MxK). Lanes 0-15: K=0..7 (v0-3), K=16..23 (v4-7);
// lanes 16-31: K=8..15, K=24..31. LDS tile [m][k], pitch LP.
__device__ __forceinline__ v16bf load_a_frag(const bf16* lds, int m0, int lane) {
  const bf16* p = lds + (m0 + (lane & 15)) * LP + ((lane >> 4) * 8);
  return frag_cat(*(const v8bf*)(p), *(const v8bf*)(p + 16));
}

// B fragment (32x16 bf16, KxN). Lanes 0-15 hold K=0..7, K=16..23 of column n; lanes 16-31 K=8..15, K=24..31.
// LDS tile stored transposed as [n][k], pitch LP.
__device__ __forceinline__ v16bf load_b_frag(const bf16* lds, int n0, int lane) {
  const bf16* p = lds + (n0 + (lane & 15)) * LP + ((lane >> 4) * 8);
  return frag_cat(*(const v8bf*)(p), *(const v8bf*)(p + 16));
}

__device__ __forceinline__ v8f wmma_bf16(v16bf a, v16bf b, v8f c) {
  return __builtin_amdgcn_wmma_f32_16x16x32_bf16(false, a, false, b, (short)0, c, false, false);
}

// Async DMA: 16B per lane, global -> LDS, tracked by ASYNCcnt (no VGPR data path).
// LDS byte offset = low 32 bits of the generic pointer (LDS aperture: addr[31:0]).
__device__ __forceinline__ void async_b128(bf16* lds_dst, const bf16* gsrc) {
  asm volatile("global_load_async_to_lds_b128 %0, %1, off"
               :
               : "v"((unsigned)(uintptr_t)lds_dst),
                 "v"((unsigned long long)(uintptr_t)gsrc)
               : "memory");
}
__device__ __forceinline__ void wait_async0() {
  asm volatile("s_wait_asynccnt 0x0" ::: "memory");
}

// ---------------- Stage 0: f32 -> bf16 (optionally exp) ----------------
__global__ void k_cvt(const float* __restrict__ src, bf16* __restrict__ dst,
                      int n4, int doExp) {
  int i = blockIdx.x * blockDim.x + threadIdx.x;
  if (i >= n4) return;
  v4f v = *(const v4f*)(src + (size_t)i * 4);
  v4bf o;
#pragma unroll
  for (int j = 0; j < 4; ++j) {
    float f = v[j];
    if (doExp) f = __expf(f);
    o[j] = (bf16)f;
  }
  *(v4bf*)(dst + (size_t)i * 4) = o;
}

// ---------------- Stage 1: fused QKV projection ----------------
// z==0: q tile -> sigmoid(q+bq) into sqT[b][d][t]
// z==1: k and v tiles (shared x tile) -> exp(k+bk), exp(k+bk)*(v+bv) into ekT/ekvT[b][d][T]
__global__ __launch_bounds__(256) void k_qkv(
    const bf16* __restrict__ xb,
    const bf16* __restrict__ Wqb, const bf16* __restrict__ Wkb, const bf16* __restrict__ Wvb,
    const float* __restrict__ bq, const float* __restrict__ bk, const float* __restrict__ bv,
    bf16* __restrict__ sqT, bf16* __restrict__ ekT, bf16* __restrict__ ekvT)
{
  __shared__ __align__(16) bf16 sA[2][128 * LP];
  __shared__ __align__(16) bf16 sB[2][2][64 * LP];

  const int tid  = threadIdx.x;
  const int lane = tid & 31;
  const int wave = tid >> 5;
  const int wm0  = (wave & 3) * 32;   // 4x2 wave grid, each wave: 32(M) x 32(N)
  const int wn0  = (wave >> 2) * 32;
  const int row0 = blockIdx.x * 128;  // flat M in [0, B*N)
  const int col0 = blockIdx.y * 64;   // output feature e/d
  const bool isQ = (blockIdx.z == 0);

  v8f acc0[2][2] = {};
  v8f acc1[2][2] = {};

  const int fr  = tid >> 1, fkc = (tid & 1) * 16;  // A fill: 128 rows x 32 halves
  const int bn  = tid >> 2, bkc = (tid & 3) * 8;   // B fill: 64 rows x 32 halves

  auto issue = [&](int buf, int K0) {
    const bf16* gA = xb + (size_t)(row0 + fr) * DD + K0 + fkc;
    async_b128(&sA[buf][fr * LP + fkc],     gA);
    async_b128(&sA[buf][fr * LP + fkc + 8], gA + 8);
    if (isQ) {
      async_b128(&sB[buf][0][bn * LP + bkc], Wqb + (size_t)(col0 + bn) * DD + K0 + bkc);
    } else {
      async_b128(&sB[buf][0][bn * LP + bkc], Wkb + (size_t)(col0 + bn) * DD + K0 + bkc);
      async_b128(&sB[buf][1][bn * LP + bkc], Wvb + (size_t)(col0 + bn) * DD + K0 + bkc);
    }
  };

  issue(0, 0);
  int buf = 0;
  for (int K0 = 0; K0 < DD; K0 += 32) {
    wait_async0();          // this wave's async copies landed in LDS
    __syncthreads();        // everyone's copies landed
    if (K0 + 32 < DD) issue(buf ^ 1, K0 + 32);   // prefetch next step into other buffer

    v16bf a0  = load_a_frag(sA[buf], wm0,      lane);
    v16bf a1  = load_a_frag(sA[buf], wm0 + 16, lane);
    v16bf b00 = load_b_frag(sB[buf][0], wn0,      lane);
    v16bf b01 = load_b_frag(sB[buf][0], wn0 + 16, lane);
    acc0[0][0] = wmma_bf16(a0, b00, acc0[0][0]);
    acc0[0][1] = wmma_bf16(a0, b01, acc0[0][1]);
    acc0[1][0] = wmma_bf16(a1, b00, acc0[1][0]);
    acc0[1][1] = wmma_bf16(a1, b01, acc0[1][1]);
    if (!isQ) {
      v16bf b10 = load_b_frag(sB[buf][1], wn0,      lane);
      v16bf b11 = load_b_frag(sB[buf][1], wn0 + 16, lane);
      acc1[0][0] = wmma_bf16(a0, b10, acc1[0][0]);
      acc1[0][1] = wmma_bf16(a0, b11, acc1[0][1]);
      acc1[1][0] = wmma_bf16(a1, b10, acc1[1][0]);
      acc1[1][1] = wmma_bf16(a1, b11, acc1[1][1]);
    }
    __syncthreads();        // all reads of this buffer done before it is refilled
    buf ^= 1;
  }

  // Epilogue: C frag element r corresponds to row m = 8*half + r, col n = lane&15.
  const int h8 = (lane >> 4) * 8;
#pragma unroll
  for (int i = 0; i < 2; ++i) {
#pragma unroll
    for (int j = 0; j < 2; ++j) {
      const int n  = col0 + wn0 + 16 * j + (lane & 15);
      const int mg = row0 + wm0 + 16 * i + h8;    // first of 8 consecutive flat rows
      const int bb = mg >> 10;                    // batch
      const int t  = mg & (NN - 1);               // position within batch
      const size_t base = ((size_t)bb * DD + n) * NN + t;   // transposed [b][d][t]
      if (isQ) {
        const float biasq = bq[n];
        v8bf o;
#pragma unroll
        for (int r = 0; r < 8; ++r) {
          float qv = acc0[i][j][r] + biasq;
          o[r] = (bf16)(1.0f / (1.0f + __expf(-qv)));
        }
        *(v8bf*)(sqT + base) = o;
      } else {
        const float biask = bk[n], biasv = bv[n];
        v8bf oek, oekv;
#pragma unroll
        for (int r = 0; r < 8; ++r) {
          float ekx = __expf(acc0[i][j][r] + biask);
          float evx = ekx * (acc1[i][j][r] + biasv);
          oek[r]  = (bf16)ekx;
          oekv[r] = (bf16)evx;
        }
        *(v8bf*)(ekT  + base) = oek;
        *(v8bf*)(ekvT + base) = oekv;
      }
    }
  }
}

// ---------------- Stage 2: num/den GEMMs + fused epilogue ----------------
// For (b, 128 t-rows, 64 d-cols): num = exp_b @ ekv, den = exp_b @ ek over T=1024,
// out = sigmoid(q) * num / den (f32), written coalesced via LDS staging.
__global__ __launch_bounds__(256) void k_aft(
    const bf16* __restrict__ ebb, const bf16* __restrict__ ekT,
    const bf16* __restrict__ ekvT, const bf16* __restrict__ sqT,
    float* __restrict__ out)
{
  __shared__ __align__(16) bf16  sA [2][128 * LP];
  __shared__ __align__(16) bf16  sBk[2][ 64 * LP];
  __shared__ __align__(16) bf16  sBv[2][ 64 * LP];
  __shared__ __align__(16) float sO [128 * 68];   // pitch 68: conflict-free staging

  const int tid  = threadIdx.x;
  const int lane = tid & 31;
  const int wave = tid >> 5;
  const int wm0  = (wave & 3) * 32;
  const int wn0  = (wave >> 2) * 32;
  const int t0 = blockIdx.x * 128;
  const int d0 = blockIdx.y * 64;
  const int b  = blockIdx.z;

  const bf16* ekb  = ekT  + (size_t)b * DD * NN;
  const bf16* ekvb = ekvT + (size_t)b * DD * NN;

  v8f cn[2][2] = {};
  v8f cd[2][2] = {};

  const int fr  = tid >> 1, fkc = (tid & 1) * 16;
  const int bn  = tid >> 2, bkc = (tid & 3) * 8;

  auto issue = [&](int buf, int K0) {
    const bf16* gA = ebb + (size_t)(t0 + fr) * NN + K0 + fkc;
    async_b128(&sA[buf][fr * LP + fkc],     gA);
    async_b128(&sA[buf][fr * LP + fkc + 8], gA + 8);
    async_b128(&sBk[buf][bn * LP + bkc], ekb  + (size_t)(d0 + bn) * NN + K0 + bkc);
    async_b128(&sBv[buf][bn * LP + bkc], ekvb + (size_t)(d0 + bn) * NN + K0 + bkc);
  };

  issue(0, 0);
  int buf = 0;
  for (int K0 = 0; K0 < NN; K0 += 32) {
    wait_async0();
    __syncthreads();
    if (K0 + 32 < NN) issue(buf ^ 1, K0 + 32);

    v16bf a0  = load_a_frag(sA[buf], wm0,      lane);
    v16bf a1  = load_a_frag(sA[buf], wm0 + 16, lane);
    v16bf bk0 = load_b_frag(sBk[buf], wn0,      lane);
    v16bf bk1 = load_b_frag(sBk[buf], wn0 + 16, lane);
    v16bf bv0 = load_b_frag(sBv[buf], wn0,      lane);
    v16bf bv1 = load_b_frag(sBv[buf], wn0 + 16, lane);
    cd[0][0] = wmma_bf16(a0, bk0, cd[0][0]);
    cd[0][1] = wmma_bf16(a0, bk1, cd[0][1]);
    cd[1][0] = wmma_bf16(a1, bk0, cd[1][0]);
    cd[1][1] = wmma_bf16(a1, bk1, cd[1][1]);
    cn[0][0] = wmma_bf16(a0, bv0, cn[0][0]);
    cn[0][1] = wmma_bf16(a0, bv1, cn[0][1]);
    cn[1][0] = wmma_bf16(a1, bv0, cn[1][0]);
    cn[1][1] = wmma_bf16(a1, bv1, cn[1][1]);
    __syncthreads();
    buf ^= 1;
  }

  const int h8 = (lane >> 4) * 8;
#pragma unroll
  for (int i = 0; i < 2; ++i) {
#pragma unroll
    for (int j = 0; j < 2; ++j) {
      const int dl = wn0 + 16 * j + (lane & 15);
      const int tl = wm0 + 16 * i + h8;           // first of 8 consecutive t
      v8bf sqv = *(const v8bf*)(sqT + ((size_t)b * DD + d0 + dl) * NN + t0 + tl);
#pragma unroll
      for (int r = 0; r < 8; ++r) {
        sO[(tl + r) * 68 + dl] = (float)sqv[r] * (cn[i][j][r] / cd[i][j][r]);
      }
    }
  }
  __syncthreads();
#pragma unroll
  for (int it = 0; it < 8; ++it) {
    int idx = tid + it * 256;        // 2048 float4 chunks of the 128x64 tile
    int row = idx >> 4;
    int c4  = (idx & 15) * 4;
    *(v4f*)(out + ((size_t)b * NN + t0 + row) * DD + d0 + c4) =
        *(const v4f*)(sO + row * 68 + c4);
  }
}

// ---------------- Host launcher ----------------
extern "C" void kernel_launch(void* const* d_in, const int* in_sizes, int n_in,
                              void* d_out, int out_size, void* d_ws, size_t ws_size,
                              hipStream_t stream)
{
  (void)in_sizes; (void)n_in; (void)out_size; (void)ws_size;
  const float* x  = (const float*)d_in[0];
  const float* Wq = (const float*)d_in[1];
  const float* bq = (const float*)d_in[2];
  const float* Wk = (const float*)d_in[3];
  const float* bk = (const float*)d_in[4];
  const float* Wv = (const float*)d_in[5];
  const float* bv = (const float*)d_in[6];
  const float* pb = (const float*)d_in[7];
  float* out = (float*)d_out;

  const size_t nx = (size_t)BB * NN * DD;  // 16,777,216
  const size_t nw = (size_t)DD * DD;       //    262,144
  const size_t nb = (size_t)NN * NN;       //  1,048,576

  char* p = (char*)d_ws;
  bf16* xb   = (bf16*)p; p += nx * 2;
  bf16* Wqb  = (bf16*)p; p += nw * 2;
  bf16* Wkb  = (bf16*)p; p += nw * 2;
  bf16* Wvb  = (bf16*)p; p += nw * 2;
  bf16* ebb  = (bf16*)p; p += nb * 2;
  bf16* sqT  = (bf16*)p; p += nx * 2;
  bf16* ekT  = (bf16*)p; p += nx * 2;
  bf16* ekvT = (bf16*)p; p += nx * 2;

  k_cvt<<<dim3((unsigned)(nx / 4 / 256)), dim3(256), 0, stream>>>(x,  xb,  (int)(nx / 4), 0);
  k_cvt<<<dim3((unsigned)(nw / 4 / 256)), dim3(256), 0, stream>>>(Wq, Wqb, (int)(nw / 4), 0);
  k_cvt<<<dim3((unsigned)(nw / 4 / 256)), dim3(256), 0, stream>>>(Wk, Wkb, (int)(nw / 4), 0);
  k_cvt<<<dim3((unsigned)(nw / 4 / 256)), dim3(256), 0, stream>>>(Wv, Wvb, (int)(nw / 4), 0);
  k_cvt<<<dim3((unsigned)(nb / 4 / 256)), dim3(256), 0, stream>>>(pb, ebb, (int)(nb / 4), 1);

  // M tiles: 32768/128 = 256; e tiles: 512/64 = 8; z: 0 = q, 1 = k+v
  k_qkv<<<dim3(256, 8, 2), dim3(256), 0, stream>>>(xb, Wqb, Wkb, Wvb, bq, bk, bv,
                                                   sqT, ekT, ekvT);
  // t tiles: 1024/128 = 8; d tiles: 512/64 = 8; batches: 32
  k_aft<<<dim3(8, 8, 32), dim3(256), 0, stream>>>(ebb, ekT, ekvT, sqT, out);
}
